// MultiHeadAttention_62843961475699
// MI455X (gfx1250) — compile-verified, matrix operand order checked by reading, not run on hardware
//
#include <hip/hip_runtime.h>
#include <hip/hip_bf16.h>

// MultiHeadAttention for MI455X (gfx1250): bf16 WMMA everywhere, flash-style
// attention (never materializes [B,H,S,S]), async (ASYNCcnt) double-buffered
// K/V staging into LDS in the attention kernel.
// B=4, S=2048, D=1024, H=16, dk=dv=64.

#define BB  4
#define SS  2048
#define DDIM 1024
#define HH  16
#define DKV 64
#define NEGV (-1.0e9f)

typedef __attribute__((ext_vector_type(16))) __bf16         v16bf;
typedef __attribute__((ext_vector_type(8)))  float          v8f;
typedef __attribute__((ext_vector_type(8)))  unsigned short v8u;
typedef __attribute__((ext_vector_type(16))) unsigned short v16u;

__device__ __forceinline__ unsigned short f32_to_bf16(float f) {
  unsigned u = __builtin_bit_cast(unsigned, f);
  u += 0x7FFFu + ((u >> 16) & 1u);          // round-to-nearest-even
  return (unsigned short)(u >> 16);
}

__device__ __forceinline__ v8f wmma_bf16(v16bf a, v16bf b, v8f c) {
  // D = A(16x32) * B(32x16) + C(16x16), f32 accumulate
  return __builtin_amdgcn_wmma_f32_16x16x32_bf16(
      false, a, false, b, (short)0, c, false, false);
}

// Load one lane's 16-element bf16 WMMA operand fragment from a K-major row.
// rowp -> k=0 of this lane's row (row = M for A, row = N for B staged
// transposed).  Fragment packing: i=0..7 -> k = 8*hi + i ;
//                                 i=8..15 -> k = 16 + 8*hi + (i-8)
// i.e. two contiguous 16-byte halves.  rowp must be 16B aligned.
__device__ __forceinline__ v16bf load_frag_row(const unsigned short* rowp, int hi) {
  v8u lo = *(const v8u*)(rowp + hi * 8);
  v8u hh = *(const v8u*)(rowp + 16 + hi * 8);
  v16u r;
#pragma unroll
  for (int i = 0; i < 8; ++i) { r[i] = lo[i]; r[i + 8] = hh[i]; }
  return __builtin_bit_cast(v16bf, r);
}

// ---- CDNA5 async global->LDS copy (ASYNCcnt-tracked), 16B per lane --------
// LDS byte address for the aperture-based generic pointer: low 32 bits
// (see ISA 10.2).
__device__ __forceinline__ unsigned lds_addr_of(const void* p) {
  return (unsigned)(size_t)p;
}
__device__ __forceinline__ void async_g2l_b128(unsigned ldsaddr,
                                               const unsigned short* gptr) {
  asm volatile("global_load_async_to_lds_b128 %0, %1, off"
               :: "v"(ldsaddr), "v"(gptr) : "memory");
}
__device__ __forceinline__ void wait_async0() {
  asm volatile("s_wait_asynccnt 0x0" ::: "memory");
}

// ---------------------------------------------------------------------------
// Kernel 1: per-head projection  Out[b,h,s,:] = In[b,s,:] @ W[h,:,:] + bias[h]
// grid = (S/128, B*H), block = 256 (8 waves, 16 output rows each)
// ---------------------------------------------------------------------------
__global__ __launch_bounds__(256) void proj_kernel(
    const float* __restrict__ In,    // [B,S,D] f32
    const float* __restrict__ W,     // [H,D,64] f32
    const float* __restrict__ bias,  // [H,64] f32
    unsigned short* __restrict__ Out)// [B,H,S,64] bf16
{
  __shared__ __align__(16) unsigned short a_lds[128 * 32]; // [row][k]
  __shared__ __align__(16) unsigned short w_lds[64 * 32];  // [n][k] (transposed)

  const int tid  = threadIdx.x;
  const int lane = tid & 31;
  const int wave = tid >> 5;
  const int hi   = lane >> 4;
  const int ln   = lane & 15;

  const int row0 = blockIdx.x * 128;
  const int bh   = blockIdx.y;
  const int h    = bh % HH;
  const int b    = bh / HH;

  const float* inB = In + (size_t)b * SS * DDIM;
  const float* Wh  = W + (size_t)h * DDIM * DKV;

  v8f acc[4];
#pragma unroll
  for (int j = 0; j < 4; ++j) acc[j] = v8f{};

  for (int kk0 = 0; kk0 < DDIM; kk0 += 32) {
    { // stage A tile 128x32, f32 -> bf16
      int r  = tid >> 1;
      int c0 = (tid & 1) * 16;
      const float4* src = (const float4*)(inB + (size_t)(row0 + r) * DDIM + kk0 + c0);
#pragma unroll
      for (int q4 = 0; q4 < 4; ++q4) {
        float4 f = src[q4];
        unsigned short* d = a_lds + r * 32 + c0 + q4 * 4;
        d[0] = f32_to_bf16(f.x); d[1] = f32_to_bf16(f.y);
        d[2] = f32_to_bf16(f.z); d[3] = f32_to_bf16(f.w);
      }
    }
    { // stage W tile 32x64, transposed to [n][k]
      int k  = tid >> 3;
      int n0 = (tid & 7) * 8;
      const float4* src = (const float4*)(Wh + (size_t)(kk0 + k) * DKV + n0);
      float4 f0 = src[0], f1 = src[1];
      w_lds[(n0 + 0) * 32 + k] = f32_to_bf16(f0.x);
      w_lds[(n0 + 1) * 32 + k] = f32_to_bf16(f0.y);
      w_lds[(n0 + 2) * 32 + k] = f32_to_bf16(f0.z);
      w_lds[(n0 + 3) * 32 + k] = f32_to_bf16(f0.w);
      w_lds[(n0 + 4) * 32 + k] = f32_to_bf16(f1.x);
      w_lds[(n0 + 5) * 32 + k] = f32_to_bf16(f1.y);
      w_lds[(n0 + 6) * 32 + k] = f32_to_bf16(f1.z);
      w_lds[(n0 + 7) * 32 + k] = f32_to_bf16(f1.w);
    }
    __syncthreads();

    v16bf afr = load_frag_row(a_lds + (wave * 16 + ln) * 32, hi);
#pragma unroll
    for (int j = 0; j < 4; ++j) {
      v16bf bfr = load_frag_row(w_lds + (j * 16 + ln) * 32, hi);
      acc[j] = wmma_bf16(afr, bfr, acc[j]);
    }
    __syncthreads();
  }

#pragma unroll
  for (int j = 0; j < 4; ++j) {
    int n = j * 16 + ln;
    float bs = bias[h * DKV + n];
#pragma unroll
    for (int r = 0; r < 8; ++r) {
      int srow = row0 + wave * 16 + r + 8 * hi;
      Out[((size_t)bh * SS + srow) * DKV + n] = f32_to_bf16(acc[j][r] + bs);
    }
  }
}

// ---------------------------------------------------------------------------
// Kernel 2: flash attention per (b,h): softmax(mask(QK^T)/8) @ V
// grid = (S/128, B*H), block = 256; wave owns 16 query rows, loops 32-key
// tiles.  K/V tiles are fetched with async global->LDS copies, double
// buffered so the next tile's HBM/L2 latency hides under this tile's WMMAs.
// ---------------------------------------------------------------------------
__global__ __launch_bounds__(256) void flash_kernel(
    const unsigned short* __restrict__ Qh,  // [B,H,S,64] bf16
    const unsigned short* __restrict__ Kh,  // [B,H,S,64] bf16
    const unsigned short* __restrict__ Vh,  // [B,H,S,64] bf16
    const int* __restrict__ mask,           // [B,S,S]
    unsigned short* __restrict__ Oc)        // [B,S,H*64] bf16 (concat layout)
{
  __shared__ __align__(16) unsigned short k_lds[2][32 * 64]; // [key][dk]
  __shared__ __align__(16) unsigned short v_raw[2][32 * 64]; // [key][dv]
  __shared__ __align__(16) unsigned short v_lds[64 * 32];    // [dv][key]
  __shared__ __align__(16) unsigned short p_lds[8][16 * 32]; // per-wave [m][key]

  const int tid  = threadIdx.x;
  const int lane = tid & 31;
  const int wave = tid >> 5;
  const int hi   = lane >> 4;
  const int ln   = lane & 15;

  const int bh = blockIdx.y;
  const int h  = bh % HH;
  const int b  = bh / HH;
  const int q0 = blockIdx.x * 128 + wave * 16;

  const unsigned short* qbase = Qh + (size_t)bh * SS * DKV;
  const unsigned short* kbase = Kh + (size_t)bh * SS * DKV;
  const unsigned short* vbase = Vh + (size_t)bh * SS * DKV;

  // Q fragments for this wave's 16 rows, whole dk=64 (2 K-steps), kept in VGPRs
  v16bf aq0 = load_frag_row(qbase + (size_t)(q0 + ln) * DKV + 0,  hi);
  v16bf aq1 = load_frag_row(qbase + (size_t)(q0 + ln) * DKV + 32, hi);

  // This thread's 16-byte slice of the 32x64 bf16 staging tiles
  const int key = tid >> 3;
  const int c0  = (tid & 7) * 8;
  unsigned kdst[2], vdst[2];
  kdst[0] = lds_addr_of(&k_lds[0][key * 64 + c0]);
  kdst[1] = lds_addr_of(&k_lds[1][key * 64 + c0]);
  vdst[0] = lds_addr_of(&v_raw[0][key * 64 + c0]);
  vdst[1] = lds_addr_of(&v_raw[1][key * 64 + c0]);

  // kick off async copies of tile 0
  async_g2l_b128(kdst[0], kbase + (size_t)key * DKV + c0);
  async_g2l_b128(vdst[0], vbase + (size_t)key * DKV + c0);

  float row_m[8], row_l[8];
#pragma unroll
  for (int r = 0; r < 8; ++r) { row_m[r] = -3.0e38f; row_l[r] = 0.f; }
  v8f acc[4];
#pragma unroll
  for (int j = 0; j < 4; ++j) acc[j] = v8f{};

  const float scale = 0.125f; // 1/sqrt(64), applied AFTER masking (as reference)

  for (int t0 = 0; t0 < SS; t0 += 32) {
    const int buf = (t0 >> 5) & 1;

    wait_async0();      // this wave's tile-t copies have landed in LDS
    __syncthreads();    // all waves: tile t ready; tile t-1 buffers now idle

    // issue async copy of tile t+1 into the free buffer (overlaps compute)
    if (t0 + 32 < SS) {
      async_g2l_b128(kdst[buf ^ 1], kbase + (size_t)(t0 + 32 + key) * DKV + c0);
      async_g2l_b128(vdst[buf ^ 1], vbase + (size_t)(t0 + 32 + key) * DKV + c0);
    }

    { // transpose V tile: v_raw[buf] ([key][dv]) -> v_lds ([dv][key])
      v8u vv = *(const v8u*)(v_raw[buf] + key * 64 + c0);
#pragma unroll
      for (int j = 0; j < 8; ++j) v_lds[(c0 + j) * 32 + key] = vv[j];
    }
    __syncthreads();

    // scores: 16 queries x 32 keys (two 16-key subtiles, K=64 in 2 WMMA each)
    v8f s[2];
#pragma unroll
    for (int ns = 0; ns < 2; ++ns) {
      v8f sc = v8f{};
      v16bf bk0 = load_frag_row(k_lds[buf] + (ns * 16 + ln) * 64 + 0,  hi);
      v16bf bk1 = load_frag_row(k_lds[buf] + (ns * 16 + ln) * 64 + 32, hi);
      sc = wmma_bf16(aq0, bk0, sc);
      sc = wmma_bf16(aq1, bk1, sc);
      s[ns] = sc;
    }

    // mask -> NEG, then scale; per-row tile max via shfl across 16-lane half
    float tmax[8];
#pragma unroll
    for (int r = 0; r < 8; ++r) {
      int qrow = q0 + r + 8 * hi;
      const int* mrow = mask + ((size_t)b * SS + qrow) * SS + t0;
      float v0 = (mrow[ln]      == 1) ? NEGV : s[0][r];
      float v1 = (mrow[16 + ln] == 1) ? NEGV : s[1][r];
      v0 *= scale; v1 *= scale;
      s[0][r] = v0; s[1][r] = v1;
      float t = fmaxf(v0, v1);
      t = fmaxf(t, __shfl_xor(t, 1, 32));
      t = fmaxf(t, __shfl_xor(t, 2, 32));
      t = fmaxf(t, __shfl_xor(t, 4, 32));
      t = fmaxf(t, __shfl_xor(t, 8, 32));
      tmax[r] = t;
    }

    // online softmax update + stash P (bf16) into per-wave LDS
#pragma unroll
    for (int r = 0; r < 8; ++r) {
      float nm    = fmaxf(row_m[r], tmax[r]);
      float alpha = __expf(row_m[r] - nm);
      float p0 = __expf(s[0][r] - nm);
      float p1 = __expf(s[1][r] - nm);
      float rs = p0 + p1;
      rs += __shfl_xor(rs, 1, 32);
      rs += __shfl_xor(rs, 2, 32);
      rs += __shfl_xor(rs, 4, 32);
      rs += __shfl_xor(rs, 8, 32);
      row_l[r] = row_l[r] * alpha + rs;
      row_m[r] = nm;
#pragma unroll
      for (int j = 0; j < 4; ++j) acc[j][r] *= alpha;
      int m = r + 8 * hi;
      p_lds[wave][m * 32 + ln]      = f32_to_bf16(p0);
      p_lds[wave][m * 32 + 16 + ln] = f32_to_bf16(p1);
    }

    // O += P(16x32) @ V(32x64)   (per-wave LDS, same-wave DS is in-order)
    v16bf ap = load_frag_row(&p_lds[wave][ln * 32], hi);
#pragma unroll
    for (int j = 0; j < 4; ++j) {
      v16bf bv = load_frag_row(v_lds + (j * 16 + ln) * 32, hi);
      acc[j] = wmma_bf16(ap, bv, acc[j]);
    }
    // next iteration's leading barrier separates these reads from buffer reuse
  }

  // normalize and store into concat layout [B,S,H*64]
#pragma unroll
  for (int j = 0; j < 4; ++j) {
#pragma unroll
    for (int r = 0; r < 8; ++r) {
      int srow = q0 + r + 8 * hi;
      float o = acc[j][r] / row_l[r];
      Oc[(((size_t)b * SS + srow) * HH + h) * DKV + j * 16 + ln] = f32_to_bf16(o);
    }
  }
}

// ---------------------------------------------------------------------------
// Kernel 3: output projection  Out = concat[B*S,1024] @ Wo[1024,1024] + bo
// grid = (B*S/128, 1024/64), block = 256
// ---------------------------------------------------------------------------
__global__ __launch_bounds__(256) void oproj_kernel(
    const unsigned short* __restrict__ Ac, // [B*S,1024] bf16
    const float* __restrict__ Wo,          // [1024,1024] f32
    const float* __restrict__ bo,          // [1024] f32
    float* __restrict__ Out)               // [B*S,1024] f32
{
  __shared__ __align__(16) unsigned short a_lds[128 * 32];
  __shared__ __align__(16) unsigned short w_lds[64 * 32];

  const int tid  = threadIdx.x;
  const int lane = tid & 31;
  const int wave = tid >> 5;
  const int hi   = lane >> 4;
  const int ln   = lane & 15;

  const int row0 = blockIdx.x * 128;
  const int n0   = blockIdx.y * 64;

  v8f acc[4];
#pragma unroll
  for (int j = 0; j < 4; ++j) acc[j] = v8f{};

  for (int kk0 = 0; kk0 < DDIM; kk0 += 32) {
    { // stage A (already bf16): straight copy
      int r  = tid >> 1;
      int c0 = (tid & 1) * 16;
      const v8u* src = (const v8u*)(Ac + (size_t)(row0 + r) * DDIM + kk0 + c0);
      *(v8u*)(a_lds + r * 32 + c0)     = src[0];
      *(v8u*)(a_lds + r * 32 + c0 + 8) = src[1];
    }
    { // stage Wo tile 32x64 transposed, f32 -> bf16
      int k  = tid >> 3;
      int c0 = (tid & 7) * 8;
      const float4* src = (const float4*)(Wo + (size_t)(kk0 + k) * DDIM + n0 + c0);
      float4 f0 = src[0], f1 = src[1];
      w_lds[(c0 + 0) * 32 + k] = f32_to_bf16(f0.x);
      w_lds[(c0 + 1) * 32 + k] = f32_to_bf16(f0.y);
      w_lds[(c0 + 2) * 32 + k] = f32_to_bf16(f0.z);
      w_lds[(c0 + 3) * 32 + k] = f32_to_bf16(f0.w);
      w_lds[(c0 + 4) * 32 + k] = f32_to_bf16(f1.x);
      w_lds[(c0 + 5) * 32 + k] = f32_to_bf16(f1.y);
      w_lds[(c0 + 6) * 32 + k] = f32_to_bf16(f1.z);
      w_lds[(c0 + 7) * 32 + k] = f32_to_bf16(f1.w);
    }
    __syncthreads();

    v16bf afr = load_frag_row(a_lds + (wave * 16 + ln) * 32, hi);
#pragma unroll
    for (int j = 0; j < 4; ++j) {
      v16bf bfr = load_frag_row(w_lds + (j * 16 + ln) * 32, hi);
      acc[j] = wmma_bf16(afr, bfr, acc[j]);
    }
    __syncthreads();
  }

#pragma unroll
  for (int j = 0; j < 4; ++j) {
    int n = n0 + j * 16 + ln;
    float bs = bo[n];
#pragma unroll
    for (int r = 0; r < 8; ++r) {
      int srow = row0 + wave * 16 + r + 8 * hi;
      Out[(size_t)srow * DDIM + n] = acc[j][r] + bs;
    }
  }
}

// ---------------------------------------------------------------------------
extern "C" void kernel_launch(void* const* d_in, const int* in_sizes, int n_in,
                              void* d_out, int out_size, void* d_ws, size_t ws_size,
                              hipStream_t stream) {
  const float* Q    = (const float*)d_in[0];
  const float* K    = (const float*)d_in[1];
  const float* V    = (const float*)d_in[2];
  const int*   mask = (const int*)d_in[3];
  const float* Wq   = (const float*)d_in[4];
  const float* bq   = (const float*)d_in[5];
  const float* Wk   = (const float*)d_in[6];
  const float* bk   = (const float*)d_in[7];
  const float* Wv   = (const float*)d_in[8];
  const float* bv   = (const float*)d_in[9];
  const float* Wo   = (const float*)d_in[10];
  const float* bo   = (const float*)d_in[11];
  float* out = (float*)d_out;

  // workspace: q/k/v heads + concat, each B*H*S*64 bf16 = 16 MB (64 MB total)
  const size_t nHead = (size_t)BB * HH * SS * DKV;
  unsigned short* qh = (unsigned short*)d_ws;
  unsigned short* kh = qh + nHead;
  unsigned short* vh = kh + nHead;
  unsigned short* oc = vh + nHead;

  dim3 blk(256);
  dim3 gproj(SS / 128, BB * HH);        // (16, 64)
  proj_kernel<<<gproj, blk, 0, stream>>>(Q, Wq, bq, qh);
  proj_kernel<<<gproj, blk, 0, stream>>>(K, Wk, bk, kh);
  proj_kernel<<<gproj, blk, 0, stream>>>(V, Wv, bv, vh);
  flash_kernel<<<gproj, blk, 0, stream>>>(qh, kh, vh, mask, oc);
  dim3 gop((BB * SS) / 128, DDIM / 64); // (64, 16)
  oproj_kernel<<<gop, blk, 0, stream>>>(oc, Wo, bo, out);
}
